// ContraNorm_81063212744819
// MI455X (gfx1250) — matrix operand order repaired and checked by reading, not compiled, not run on hardware
//
#include <hip/hip_runtime.h>
#include <hip/hip_bf16.h>

// ContraNorm as flash-attention on CDNA5 (gfx1250, wave32, WMMA f16->f32).
// out = 1.1*x - 0.1 * softmax(xn @ xn^T) @ x,  N=16384, D=256.
// Scores are cosines in [-1,1] => exp(s) never overflows => no online max
// tracking / accumulator rescale needed (softmax is shift-invariant).

typedef _Float16 v16h __attribute__((ext_vector_type(16)));
typedef _Float16 v8h  __attribute__((ext_vector_type(8)));
typedef float    v8f  __attribute__((ext_vector_type(8)));

#define N_TOK 16384
#define DIM   256
#define WAVES 2          // waves per workgroup; each wave owns 32 output rows
#define LOG2E 1.44269504088896340736f

// ---------------------------------------------------------------------------
// Kernel 1: Q = f16( x / max(||x||_2, eps) ), one row per 256-thread block.
// ---------------------------------------------------------------------------
__global__ void prep_q(const float* __restrict__ x, _Float16* __restrict__ q) {
    const int row  = blockIdx.x;
    const int t    = threadIdx.x;
    const int lane = t & 31;
    const int wid  = t >> 5;

    float v  = x[(size_t)row * DIM + t];
    float ss = v * v;
    #pragma unroll
    for (int m = 16; m >= 1; m >>= 1) ss += __shfl_xor(ss, m, 32);

    __shared__ float acc[8];
    if (lane == 0) acc[wid] = ss;
    __syncthreads();
    float tot = 0.f;
    #pragma unroll
    for (int i = 0; i < 8; ++i) tot += acc[i];

    float inv = 1.0f / fmaxf(sqrtf(tot), 1e-12f);
    q[(size_t)row * DIM + t] = (_Float16)(v * inv);
}

// ---------------------------------------------------------------------------
// Kernel 2: Vt[f][tok] = f16(x[tok][f])  (256 x 16384), LDS-tiled transpose.
// Makes PV B-fragments contiguous 32B per-lane global loads.
// ---------------------------------------------------------------------------
__global__ void prep_vt(const float* __restrict__ x, _Float16* __restrict__ vt) {
    __shared__ _Float16 tile[32][33];   // +1 to dodge bank conflicts
    const int tx   = threadIdx.x;       // 0..31
    const int ty   = threadIdx.y;       // 0..7
    const int tok0 = blockIdx.x * 32;
    const int ft0  = blockIdx.y * 32;

    #pragma unroll
    for (int k = 0; k < 4; ++k) {
        int tok = tok0 + ty + k * 8;
        int ft  = ft0 + tx;
        tile[ty + k * 8][tx] = (_Float16)x[(size_t)tok * DIM + ft];
    }
    __syncthreads();
    #pragma unroll
    for (int k = 0; k < 4; ++k) {
        int ft  = ft0 + ty + k * 8;
        int tok = tok0 + tx;
        vt[(size_t)ft * N_TOK + tok] = tile[tx][ty + k * 8];
    }
}

// ---------------------------------------------------------------------------
// Kernel 3: flash attention. Each wave: 32 rows (two 16-row A tiles); every
// loaded K/V B-fragment feeds TWO WMMAs. Per 32-key iteration, bulk-phase
// structure for deep memory-level parallelism (single wave per SIMD):
//   phase 1: issue ALL 16 K-fragment loads, then 32 S-WMMAs
//   phase 2: issue ALL 16 V-fragment loads (latency hidden by softmax + LDS)
//   phase 3: exp(S) -> P staged f16 in per-wave LDS, reread in A layout
//   phase 4: 32 PV-WMMAs on landed V fragments
// ---------------------------------------------------------------------------
__global__ __launch_bounds__(WAVES * 32, 1)
void attn(const float* __restrict__ x,
          const _Float16* __restrict__ q,
          const _Float16* __restrict__ vt,
          float* __restrict__ out) {
    const int tid   = threadIdx.x;
    const int lane  = tid & 31;
    const int wid   = tid >> 5;
    const int mlane = lane & 15;   // row (A) or col (B/C) within 16
    const int khalf = lane >> 4;   // which K-half of the fragment this lane holds
    const int mbase = (blockIdx.x * WAVES + wid) * 32;

    __shared__ __align__(32) _Float16 pbuf[WAVES][2][16 * 32];

    // --- resident Q A-fragments: 2 M-tiles x 8 chunks of 16x32 (128 VGPRs) ---
    v16h a[2][8];
    #pragma unroll
    for (int j = 0; j < 2; ++j) {
        const _Float16* qr = q + (size_t)(mbase + j * 16 + mlane) * DIM;
        #pragma unroll
        for (int c = 0; c < 8; ++c) {
            v8h lo = *(const v8h*)(qr + c * 32 + khalf * 16);       // K = kb+16h+0..7
            v8h hi = *(const v8h*)(qr + c * 32 + khalf * 16 + 8);   // K = kb+16h+8..15
            a[j][c] = __builtin_shufflevector(lo, hi, 0,1,2,3,4,5,6,7,
                                                      8,9,10,11,12,13,14,15);
        }
    }

    // --- accumulators --------------------------------------------------------
    v8f o[2][16];
    #pragma unroll
    for (int j = 0; j < 2; ++j)
        #pragma unroll
        for (int t = 0; t < 16; ++t)
            #pragma unroll
            for (int r = 0; r < 8; ++r) o[j][t][r] = 0.f;

    float sm[2][8];
    #pragma unroll
    for (int j = 0; j < 2; ++j)
        #pragma unroll
        for (int r = 0; r < 8; ++r) sm[j][r] = 0.f;

    // --- main loop over key blocks of 32 ------------------------------------
    for (int k0 = 0; k0 < N_TOK; k0 += 32) {
        __builtin_prefetch(q + (size_t)(k0 + 32) * DIM, 0, 1); // next key block

        // Phase 1a: issue ALL K-fragment loads for this block (16 b128-pairs)
        const _Float16* kr0 = q + (size_t)(k0 + mlane) * DIM + khalf * 16;
        const _Float16* kr1 = kr0 + (size_t)16 * DIM;
        v16h kb0[8], kb1[8];
        #pragma unroll
        for (int c = 0; c < 8; ++c) {
            kb0[c] = *(const v16h*)(kr0 + c * 32);
            kb1[c] = *(const v16h*)(kr1 + c * 32);
        }

        // Phase 1b: S = Q @ K^T : 2x2 tiles, reduce over D in 8 chunks of 32
        v8f s[2][2];
        #pragma unroll
        for (int j = 0; j < 2; ++j)
            #pragma unroll
            for (int r = 0; r < 8; ++r) { s[j][0][r] = 0.f; s[j][1][r] = 0.f; }

        #pragma unroll
        for (int c = 0; c < 8; ++c) {
            s[0][0] = __builtin_amdgcn_wmma_f32_16x16x32_f16(
                          false, a[0][c], false, kb0[c], (short)0, s[0][0], false, false);
            s[1][0] = __builtin_amdgcn_wmma_f32_16x16x32_f16(
                          false, a[1][c], false, kb0[c], (short)0, s[1][0], false, false);
            s[0][1] = __builtin_amdgcn_wmma_f32_16x16x32_f16(
                          false, a[0][c], false, kb1[c], (short)0, s[0][1], false, false);
            s[1][1] = __builtin_amdgcn_wmma_f32_16x16x32_f16(
                          false, a[1][c], false, kb1[c], (short)0, s[1][1], false, false);
        }

        // Phase 2: issue ALL V-fragment loads now; their latency is hidden
        // behind the softmax VALU work and the LDS P round-trip below.
        v16h bv[16];
        #pragma unroll
        for (int t = 0; t < 16; ++t)
            bv[t] = *(const v16h*)(vt + (size_t)(t * 16 + mlane) * N_TOK
                                      + k0 + khalf * 16);

        // Phase 3: P = exp(S). Scores are cosines in [-1,1] -> no max needed.
        // C layout: reg r holds row (mbase + j*16 + r + 8*khalf), col = mlane.
        #pragma unroll
        for (int j = 0; j < 2; ++j) {
            _Float16* pb = &pbuf[wid][j][0];
            #pragma unroll
            for (int r = 0; r < 8; ++r) {
                float p0 = exp2f(s[j][0][r] * LOG2E);
                float p1 = exp2f(s[j][1][r] * LOG2E);
                float rs = p0 + p1;
                rs += __shfl_xor(rs, 1, 32);
                rs += __shfl_xor(rs, 2, 32);
                rs += __shfl_xor(rs, 4, 32);
                rs += __shfl_xor(rs, 8, 32);
                sm[j][r] += rs;
                pb[(r + 8 * khalf) * 32 + mlane]      = (_Float16)p0;
                pb[(r + 8 * khalf) * 32 + 16 + mlane] = (_Float16)p1;
            }
        }
        asm volatile("s_wait_dscnt 0x0" ::: "memory");

        // Re-read P in A-fragment layout (per-wave LDS, in-order DS)
        v16h pa[2];
        #pragma unroll
        for (int j = 0; j < 2; ++j) {
            const _Float16* pb = &pbuf[wid][j][0];
            v8h lo = *(const v8h*)(pb + mlane * 32 + khalf * 16);
            v8h hi = *(const v8h*)(pb + mlane * 32 + khalf * 16 + 8);
            pa[j] = __builtin_shufflevector(lo, hi, 0,1,2,3,4,5,6,7,
                                                    8,9,10,11,12,13,14,15);
        }

        // Phase 4: O += P @ V on landed fragments, each bv reused 2x
        #pragma unroll
        for (int t = 0; t < 16; ++t) {
            o[0][t] = __builtin_amdgcn_wmma_f32_16x16x32_f16(
                          false, pa[0], false, bv[t], (short)0, o[0][t], false, false);
            o[1][t] = __builtin_amdgcn_wmma_f32_16x16x32_f16(
                          false, pa[1], false, bv[t], (short)0, o[1][t], false, false);
        }
    }

    // --- epilogue: out = 1.1*x - 0.1 * O/rowsum ------------------------------
    #pragma unroll
    for (int j = 0; j < 2; ++j) {
        float inv[8];
        #pragma unroll
        for (int r = 0; r < 8; ++r) inv[r] = 1.0f / sm[j][r];
        #pragma unroll
        for (int t = 0; t < 16; ++t) {
            #pragma unroll
            for (int r = 0; r < 8; ++r) {
                int m = mbase + j * 16 + r + 8 * khalf;
                int n = t * 16 + mlane;
                float val = o[j][t][r] * inv[r];
                out[(size_t)m * DIM + n] =
                    1.1f * x[(size_t)m * DIM + n] - 0.1f * val;
            }
        }
    }
}

// ---------------------------------------------------------------------------
extern "C" void kernel_launch(void* const* d_in, const int* in_sizes, int n_in,
                              void* d_out, int out_size, void* d_ws, size_t ws_size,
                              hipStream_t stream) {
    const float* x = (const float*)d_in[0];
    float* out = (float*)d_out;

    _Float16* q  = (_Float16*)d_ws;                                    // 8 MB
    _Float16* vt = (_Float16*)((char*)d_ws + (size_t)N_TOK * DIM * 2); // 8 MB

    prep_q <<<N_TOK, 256, 0, stream>>>(x, q);
    prep_vt<<<dim3(N_TOK / 32, DIM / 32), dim3(32, 8), 0, stream>>>(x, vt);
    attn   <<<N_TOK / (32 * WAVES), WAVES * 32, 0, stream>>>(x, q, vt, out);
}
